// HRVQ_16621523435916
// MI455X (gfx1250) — compile-verified, matrix operand order checked by reading, not run on hardware
//
#include <hip/hip_runtime.h>
#include <math.h>

typedef __attribute__((ext_vector_type(16))) __bf16 v16bf;
typedef __attribute__((ext_vector_type(8)))  float  v8f;
typedef __attribute__((ext_vector_type(4)))  unsigned int v4u;
typedef __attribute__((ext_vector_type(8)))  int    v8i;
typedef __attribute__((ext_vector_type(4)))  int    v4i;

#define EMBED_DIM 256
#define NUM_CODES 512
#define M_TILE    128
#define RS        260   // residual LDS row stride (floats), breaks bank conflicts
#define BS        272   // B-tile LDS row stride (bf16 elems), breaks bank conflicts

#if __has_builtin(__builtin_amdgcn_tensor_load_to_lds)
#define HAS_TDM 1
#else
#define HAS_TDM 0
#endif

#if __has_builtin(__builtin_amdgcn_sched_group_barrier)
#define SGB(mask, cnt) __builtin_amdgcn_sched_group_barrier((mask), (cnt), 0)
#else
#define SGB(mask, cnt)
#endif

// ---------------------------------------------------------------------------
// Prep: convert codebooks fp32 -> bf16, compute ||e||^2, zero counters.
// ---------------------------------------------------------------------------
__global__ void hrvq_prep(const float* __restrict__ e0, const float* __restrict__ e1,
                          const float* __restrict__ e2,
                          unsigned short* __restrict__ embbf, float* __restrict__ enorm,
                          float* __restrict__ counts, float* __restrict__ sumsq) {
    int gid = blockIdx.x * 256 + threadIdx.x;
    if (gid < 3 * NUM_CODES) {
        int l = gid >> 9;
        int c = gid & (NUM_CODES - 1);
        const float* src = ((l == 0) ? e0 : (l == 1) ? e1 : e2) + (size_t)c * EMBED_DIM;
        __bf16* dst = (__bf16*)(embbf + (size_t)gid * EMBED_DIM);
        float s = 0.0f;
        for (int k = 0; k < EMBED_DIM; ++k) {
            float f = src[k];
            s += f * f;
            dst[k] = (__bf16)f;
        }
        enorm[gid] = s;
    }
    if (blockIdx.x == 6) {
        for (int i = threadIdx.x; i < 3 * NUM_CODES; i += 256) counts[i] = 0.0f;
        if (threadIdx.x < 3) sumsq[threadIdx.x] = 0.0f;
    }
}

#if HAS_TDM
// Issue a TDM DMA: 32 codes x 256 bf16 (row stride 256) -> LDS with 8-DWORD pad
// after every 128 DWORDs (i.e. LDS row stride 272 bf16 = BS).
__device__ __forceinline__ void tdm_load_btile(const unsigned short* gsrc,
                                               unsigned int lds_off) {
    unsigned long long ga = (unsigned long long)(uintptr_t)gsrc;
    v4u g0;
    g0[0] = 1u;                                            // count=1, user descriptor
    g0[1] = lds_off;                                       // lds_addr
    g0[2] = (unsigned int)(ga & 0xFFFFFFFFu);              // global_addr[31:0]
    g0[3] = (unsigned int)((ga >> 32) & 0x01FFFFFFu) | (2u << 30);  // addr[56:32] | type=2
    v8i g1;
    g1[0] = (1 << 16)        // data_size = 2 bytes
          | (1 << 20)        // pad_enable
          | (6 << 22)        // pad_interval: 128 DWORDs
          | (7 << 25);       // pad_amount: 8 DWORDs (32 B) -> row stride 544 B
    g1[1] = (256 << 16);     // tensor_dim0 = 256 elems (bits 79:48, low half)
    g1[2] = (512 << 16);     // tensor_dim0 hi = 0 ; tensor_dim1 lo = 512 rows
    g1[3] = (256 << 16);     // tensor_dim1 hi = 0 ; tile_dim0 = 256
    g1[4] = 32;              // tile_dim1 = 32 rows ; tile_dim2 = 0
    g1[5] = 256;             // tensor_dim0_stride = 256 elems (lo 32)
    g1[6] = 0;               // stride hi ; tensor_dim1_stride lo (unused for 2D)
    g1[7] = 0;
    v4i gz4 = {0, 0, 0, 0};
    v8i gz8 = {0, 0, 0, 0, 0, 0, 0, 0};
    __builtin_amdgcn_tensor_load_to_lds(g0, g1, gz4, gz4, gz8, 0);
}
#endif

// ---------------------------------------------------------------------------
// Main: per WG handle 128 rows. 3-level residual VQ with bf16 WMMA distances.
// ---------------------------------------------------------------------------
__global__ __launch_bounds__(256, 1)
void hrvq_main(const float* __restrict__ z,
               const float* __restrict__ e0, const float* __restrict__ e1,
               const float* __restrict__ e2,
               const unsigned short* __restrict__ embbf,
               const float* __restrict__ enorm,
               float* __restrict__ counts, float* __restrict__ sumsq,
               float* __restrict__ zq_out, float* __restrict__ idx_out, int N) {
    __shared__ float res[M_TILE * RS];                 // fp32 residual, 133120 B
    __shared__ unsigned short Bt[2][32 * BS];          // double-buffered bf16 codes
    __shared__ int idxbuf[M_TILE];

    const int tid  = threadIdx.x;
    const int wave = tid >> 5;
    const int lane = tid & 31;
    const int hh   = lane >> 4;        // half of wave (0: lanes 0-15, 1: lanes 16-31)
    const int ln   = lane & 15;
    const int g    = blockIdx.x * M_TILE;

    // --- load residual = z ---
    for (int i = tid; i < M_TILE * 64; i += 256) {     // float4 granules
        int r = i >> 6, c4 = (i & 63) << 2;
        float4 v = *(const float4*)(z + (size_t)(g + r) * EMBED_DIM + c4);
        *(float4*)(&res[r * RS + c4]) = v;
    }
    __syncthreads();

    for (int l = 0; l < 3; ++l) {
        const unsigned short* ebf = embbf + (size_t)l * NUM_CODES * EMBED_DIM;

#if HAS_TDM
        if (wave == 0)
            tdm_load_btile(ebf, (unsigned int)(uintptr_t)&Bt[0][0]);
#endif
        // --- A fragments for this wave's 16 rows (bf16, f32-accum WMMA layout) ---
        v16bf a[8];
        {
            int m = wave * 16 + ln;
            #pragma unroll
            for (int ch = 0; ch < 8; ++ch) {
                int kb = ch * 32 + hh * 8;             // lanes<16: K 0..7,16..23 ; lanes>=16: +8
                const float* p = &res[m * RS + kb];
                union { __bf16 h[16]; v16bf v; } u;
                #pragma unroll
                for (int q = 0; q < 8; ++q) u.h[q] = (__bf16)p[q];
                #pragma unroll
                for (int q = 0; q < 8; ++q) u.h[8 + q] = (__bf16)p[16 + q];
                a[ch] = u.v;
            }
        }
        float minv[8];
        int   mini[8];
        #pragma unroll
        for (int j = 0; j < 8; ++j) { minv[j] = 3.4e38f; mini[j] = 0; }

        int cur = 0;
        for (int ph = 0; ph < 16; ++ph) {              // 32 codes per phase
#if HAS_TDM
            if (wave == 0) __builtin_amdgcn_s_wait_tensorcnt((short)0);
            __syncthreads();                           // Bt[cur] ready for everyone
            if (wave == 0 && ph < 15)
                tdm_load_btile(ebf + (size_t)(ph + 1) * 32 * EMBED_DIM,
                               (unsigned int)(uintptr_t)&Bt[cur ^ 1][0]);
#else
            // manual staging into Bt[cur]
            for (int i = tid; i < 32 * 32; i += 256) { // uint4 granules (8 bf16)
                int c = i >> 5, k8 = (i & 31) << 3;
                uint4 v = *(const uint4*)(ebf + (size_t)(ph * 32 + c) * EMBED_DIM + k8);
                *(uint4*)(&Bt[cur][c * BS + k8]) = v;
            }
            __syncthreads();
#endif
            const unsigned short* Bb = &Bt[cur][0];
            const int cc0 = ln;                        // tile-0 code row in Bt
            const int cc1 = 16 + ln;                   // tile-1 code row in Bt
            const int code0 = ph * 32 + ln;
            const int code1 = ph * 32 + 16 + ln;
            float en0 = enorm[l * NUM_CODES + code0];
            float en1 = enorm[l * NUM_CODES + code1];

            // software-pipelined B fragments: double-buffered regs per tile so
            // chunk ch+1 LDS loads stay outstanding during chunk ch WMMAs.
            union Bf { uint4 u[2]; v16bf v; };
            Bf b0[2], b1[2];
            {
                int kb = hh * 8;
                b0[0].u[0] = *(const uint4*)(&Bb[cc0 * BS + kb]);
                b0[0].u[1] = *(const uint4*)(&Bb[cc0 * BS + kb + 16]);
                b1[0].u[0] = *(const uint4*)(&Bb[cc1 * BS + kb]);
                b1[0].u[1] = *(const uint4*)(&Bb[cc1 * BS + kb + 16]);
            }
            v8f acc0 = {0.f, 0.f, 0.f, 0.f, 0.f, 0.f, 0.f, 0.f};
            v8f acc1 = {0.f, 0.f, 0.f, 0.f, 0.f, 0.f, 0.f, 0.f};
            #pragma unroll
            for (int ch = 0; ch < 8; ++ch) {
                const int cb = ch & 1, nb = cb ^ 1;
                if (ch < 7) {
                    int kb = (ch + 1) * 32 + hh * 8;
                    b0[nb].u[0] = *(const uint4*)(&Bb[cc0 * BS + kb]);
                    b0[nb].u[1] = *(const uint4*)(&Bb[cc0 * BS + kb + 16]);
                    b1[nb].u[0] = *(const uint4*)(&Bb[cc1 * BS + kb]);
                    b1[nb].u[1] = *(const uint4*)(&Bb[cc1 * BS + kb + 16]);
                }
                acc0 = __builtin_amdgcn_wmma_f32_16x16x32_bf16(
                    false, a[ch], false, b0[cb].v, (short)0, acc0, false, false);
                acc1 = __builtin_amdgcn_wmma_f32_16x16x32_bf16(
                    false, a[ch], false, b1[cb].v, (short)0, acc1, false, false);
            }
            // Scheduling shape: DS4[ch0], then {DS4[ch+1], WMMA2[ch]} x7, WMMA2[ch7]
            // -> chunk ch+1 loads are issued before the WMMAs consuming chunk ch,
            //    so 4 LDS loads stay outstanding under every WMMA pair.
            SGB(0x100, 4);                 // DS reads: ch0
            #pragma unroll
            for (int it = 0; it < 7; ++it) {
                SGB(0x100, 4);             // DS reads: ch(it+1)
                SGB(0x008, 2);             // WMMA pair: ch(it)
            }
            SGB(0x008, 2);                 // WMMA pair: ch7

            // per-lane running argmin (cross-lane reduction deferred to level end)
            #pragma unroll
            for (int j = 0; j < 8; ++j) {
                float v0 = fmaf(-2.0f, acc0[j], en0);  // ||e||^2 - 2 r.e
                bool c0 = v0 < minv[j];
                minv[j] = c0 ? v0 : minv[j];
                mini[j] = c0 ? code0 : mini[j];
                float v1 = fmaf(-2.0f, acc1[j], en1);
                bool c1 = v1 < minv[j];
                minv[j] = c1 ? v1 : minv[j];
                mini[j] = c1 ? code1 : mini[j];
            }
            cur ^= 1;
#if !HAS_TDM
            __syncthreads();
#endif
        }

        // --- one cross-lane argmin reduction per level (within 16-lane halves) ---
        #pragma unroll
        for (int j = 0; j < 8; ++j) {
            float v = minv[j];
            int  ci = mini[j];
            #pragma unroll
            for (int ms = 1; ms < 16; ms <<= 1) {
                float ov = __shfl_xor(v, ms, 32);
                int   oi = __shfl_xor(ci, ms, 32);
                if (ov < v) { v = ov; ci = oi; }
            }
            minv[j] = v; mini[j] = ci;
        }

        // --- publish indices (lanes 0 and 16 hold rows j and j+8) ---
        if (ln == 0) {
            #pragma unroll
            for (int j = 0; j < 8; ++j) {
                int r = wave * 16 + hh * 8 + j;
                idxbuf[r] = mini[j];
                idx_out[(size_t)l * N + g + r] = (float)mini[j];
                atomicAdd(&counts[l * NUM_CODES + mini[j]], 1.0f);
            }
        }
        __syncthreads();

        // --- residual -= code ; accumulate ||diff||^2 (diff == new residual) ---
        const float* el = (l == 0) ? e0 : (l == 1) ? e1 : e2;
        float ls = 0.0f;
        for (int e = tid; e < M_TILE * EMBED_DIM; e += 256) {
            int r = e >> 8, c = e & 255;
            float q  = el[(size_t)idxbuf[r] * EMBED_DIM + c];
            float nv = res[r * RS + c] - q;
            res[r * RS + c] = nv;
            ls += nv * nv;
        }
        #pragma unroll
        for (int ms = 16; ms >= 1; ms >>= 1) ls += __shfl_xor(ls, ms, 32);
        if (lane == 0) atomicAdd(&sumsq[l], ls);
        __syncthreads();
    }

    // --- z_q_st = z + (z_q_sum - z) = z - residual_final ---
    for (int i = tid; i < M_TILE * 64; i += 256) {
        int r = i >> 6, c4 = (i & 63) << 2;
        float4 zv = *(const float4*)(z + (size_t)(g + r) * EMBED_DIM + c4);
        float4 rv = *(const float4*)(&res[r * RS + c4]);
        float4 o;
        o.x = zv.x - rv.x; o.y = zv.y - rv.y; o.z = zv.z - rv.z; o.w = zv.w - rv.w;
        *(float4*)(zq_out + (size_t)(g + r) * EMBED_DIM + c4) = o;
    }
}

// ---------------------------------------------------------------------------
// Final: vq_loss and 3 perplexities.
// ---------------------------------------------------------------------------
__global__ void hrvq_final(const float* __restrict__ counts, const float* __restrict__ sumsq,
                           float* __restrict__ out_scalars, float invN, float invND) {
    __shared__ float red[NUM_CODES];
    int t = threadIdx.x;
    for (int l = 0; l < 3; ++l) {
        float p = counts[l * NUM_CODES + t] * invN;
        red[t] = p * logf(p + 1e-10f);
        __syncthreads();
        for (int s = NUM_CODES / 2; s >= 1; s >>= 1) {
            if (t < s) red[t] += red[t + s];
            __syncthreads();
        }
        if (t == 0) out_scalars[1 + l] = expf(-red[0]);
        __syncthreads();
    }
    if (t == 0)
        out_scalars[0] = (0.25f * sumsq[0] + 0.5f * sumsq[1] + 1.0f * sumsq[2]) * invND;
}

extern "C" void kernel_launch(void* const* d_in, const int* in_sizes, int n_in,
                              void* d_out, int out_size, void* d_ws, size_t ws_size,
                              hipStream_t stream) {
    const float* z  = (const float*)d_in[0];
    const float* e0 = (const float*)d_in[1];
    const float* e1 = (const float*)d_in[2];
    const float* e2 = (const float*)d_in[3];
    const int N = in_sizes[0] / EMBED_DIM;           // 262144 rows

    float* out = (float*)d_out;
    char*  ws  = (char*)d_ws;
    float* counts = (float*)ws;                      // 3*512 f32   = 6144 B
    float* sq     = (float*)(ws + 6144);             // 3 f32 (pad) @6144
    float* enorm  = (float*)(ws + 6176);             // 3*512 f32   @6176
    unsigned short* embbf = (unsigned short*)(ws + 6176 + 6144);  // 768 KB bf16 codebooks

    hrvq_prep<<<7, 256, 0, stream>>>(e0, e1, e2, embbf, enorm, counts, sq);

    int nblocks = N / M_TILE;                        // 2048
    hrvq_main<<<nblocks, 256, 0, stream>>>(z, e0, e1, e2, embbf, enorm, counts, sq,
                                           out, out + (size_t)N * EMBED_DIM, N);

    hrvq_final<<<1, NUM_CODES, 0, stream>>>(counts, sq,
                                            out + (size_t)N * EMBED_DIM + 3 * (size_t)N,
                                            1.0f / (float)N,
                                            1.0f / ((float)N * (float)EMBED_DIM));
}
